// Head_3195455668854
// MI455X (gfx1250) — matrix operand order repaired and checked by reading, not compiled, not run on hardware
//
#include <hip/hip_runtime.h>

#define TT 2048
#define CC 1024
#define HH 128
#define BB 8

typedef __attribute__((ext_vector_type(16))) __bf16 v16bf;
typedef __attribute__((ext_vector_type(8)))  __bf16 v8bf;
typedef __attribute__((ext_vector_type(8)))  float  v8f;
typedef __attribute__((ext_vector_type(4)))  float  v4f;

// ---------------------------------------------------------------------------
// Kernel 0: transpose + convert a [C,H] fp32 weight matrix to [H,C] bf16.
// ---------------------------------------------------------------------------
__global__ __launch_bounds__(256) void wtrans_kernel(const float* __restrict__ W,
                                                     __bf16* __restrict__ Wt) {
  int idx = blockIdx.x * 256 + threadIdx.x;   // 1024*128 = 131072 elements
  int h = idx & (HH - 1);
  int c = idx >> 7;
  Wt[(size_t)h * CC + c] = (__bf16)W[(size_t)c * HH + h];
}

// --------------------------- fragment helpers ------------------------------
__device__ __forceinline__ void load_b8(v16bf (&B)[8],
                                        const __bf16* __restrict__ wbase,
                                        int k0) {
#pragma unroll
  for (int nt = 0; nt < 8; ++nt)
    B[nt] = *(const v16bf*)(wbase + (size_t)nt * 16 * CC + k0);
}

__device__ __forceinline__ void load_a4(v4f (&a)[4],
                                        const float* __restrict__ xrow,
                                        int k0) {
  a[0] = *(const v4f*)(xrow + k0);
  a[1] = *(const v4f*)(xrow + k0 + 4);
  a[2] = *(const v4f*)(xrow + k0 + 8);
  a[3] = *(const v4f*)(xrow + k0 + 12);
}

__device__ __forceinline__ v16bf cvt_a(const v4f (&a)[4]) {
  v16bf A;
#pragma unroll
  for (int i = 0; i < 4; ++i) {
    A[i]      = (__bf16)a[0][i];
    A[i + 4]  = (__bf16)a[1][i];
    A[i + 8]  = (__bf16)a[2][i];
    A[i + 12] = (__bf16)a[3][i];
  }
  return A;
}

__device__ __forceinline__ void wmma8(v8f (&acc)[8], v16bf A,
                                      const v16bf (&B)[8]) {
#pragma unroll
  for (int nt = 0; nt < 8; ++nt)
    acc[nt] = __builtin_amdgcn_wmma_f32_16x16x32_bf16(false, A, false, B[nt],
                                                      (short)0, acc[nt],
                                                      false, false);
}

__device__ __forceinline__ void load_k4(v16bf (&B)[4],
                                        const __bf16* __restrict__ kp, int s) {
#pragma unroll
  for (int c = 0; c < 4; ++c)
    B[c] = *(const v16bf*)(kp + (size_t)(16 * s) * HH + 32 * c);
}

__device__ __forceinline__ v8f wmma4(v8f S, const v16bf (&Aq)[4],
                                     const v16bf (&B)[4]) {
#pragma unroll
  for (int c = 0; c < 4; ++c)
    S = __builtin_amdgcn_wmma_f32_16x16x32_bf16(false, Aq[c], false, B[c],
                                                (short)0, S, false, false);
  return S;
}

__device__ __forceinline__ void load_v8(v16bf (&B)[8],
                                        const __bf16* __restrict__ vb) {
#pragma unroll
  for (int c = 0; c < 8; ++c)
    B[c] = *(const v16bf*)(vb + (size_t)(16 * c) * TT);
}

// ---------------------------------------------------------------------------
// Kernel 1: projections Q/K/V = X @ W as bf16 WMMA GEMMs, software-pipelined.
// One wave computes a 16x128 strip. B fragments are double-buffered across
// 32-deep K-stages (B0/B1 in distinct registers): each stage's WMMA chain
// runs against fragments loaded a stage earlier, while the next stage's 16
// b128 loads are in flight. A loads are issued BEFORE the long B-load group
// so waiting on A (in-order LOADcnt) never drains the fresh B loads.
// grid = (B*T/16, 3), block = 32 (one wave).
// ---------------------------------------------------------------------------
__global__ __launch_bounds__(32) void proj_kernel(
    const float* __restrict__ X,
    const __bf16* __restrict__ Wtq, const __bf16* __restrict__ Wtk,
    const __bf16* __restrict__ Wtv,
    __bf16* __restrict__ Qb, __bf16* __restrict__ Kb, __bf16* __restrict__ Vt) {
  const int mt = blockIdx.x, w = blockIdx.y;
  const int lane = threadIdx.x & 31, half = lane >> 4, l16 = lane & 15;
  const __bf16* Wt = (w == 0) ? Wtq : (w == 1) ? Wtk : Wtv;
  const int m0 = mt * 16;

  v8f acc[8];
#pragma unroll
  for (int nt = 0; nt < 8; ++nt)
#pragma unroll
    for (int r = 0; r < 8; ++r) acc[nt][r] = 0.f;

  const float*  xrow  = X  + (size_t)(m0 + l16) * CC + 16 * half;
  const __bf16* wbase = Wt + (size_t)l16 * CC + 16 * half;

  v16bf B0[8], B1[8];
  load_b8(B0, wbase, 0);                       // prologue: stage-0 fragments

  for (int k0 = 0; k0 < CC; k0 += 64) {
    v4f a[4];
    // Stage 0: consume B0 (resident), keep B1 loads in flight.
    load_a4(a, xrow, k0);                      // 4 loads (short group first)
    load_b8(B1, wbase, k0 + 32);               // 16 loads for next stage
    wmma8(acc, cvt_a(a), B0);                  // waits only on the A group

    // Stage 1: consume B1, refill B0 for the next iteration.
    load_a4(a, xrow, k0 + 32);
    if (k0 + 64 < CC) load_b8(B0, wbase, k0 + 64);
    wmma8(acc, cvt_a(a), B1);
  }

  // C/D layout: VGPR r, lanes 0-15 -> (M=r, N=lane), lanes 16-31 -> (M=8+r).
  if (w < 2) {
    __bf16* Outp = (w == 0) ? Qb : Kb;
#pragma unroll
    for (int nt = 0; nt < 8; ++nt)
#pragma unroll
      for (int r = 0; r < 8; ++r)
        Outp[(size_t)(m0 + r + 8 * half) * HH + nt * 16 + l16] =
            (__bf16)acc[nt][r];
  } else {
    // V transposed: Vt[b][h][t]; per lane 8 consecutive t -> one 16B store.
    const int b = m0 >> 11;            // /2048
    const int t_in = m0 & (TT - 1);
#pragma unroll
    for (int nt = 0; nt < 8; ++nt) {
      v8bf pk;
#pragma unroll
      for (int r = 0; r < 8; ++r) pk[r] = (__bf16)acc[nt][r];
      *(v8bf*)(Vt + ((size_t)b * HH + nt * 16 + l16) * TT + t_in + 8 * half) = pk;
    }
  }
}

// ---------------------------------------------------------------------------
// Kernel 2: fused causal flash attention. One wave owns a 16-row query tile,
// streams 64-key blocks (<=16 S-WMMAs + 16 PV-WMMAs per online-softmax pass).
// K fragments double-buffer across 16-key subtiles; the g=0 V fragments are
// loaded BEFORE the softmax so exp/shfl VALU work hides their latency.
// Fully-masked subtiles are skipped via the wave-uniform `na` count, which
// also keeps every K/V load in-bounds (j <= T-64 always).
// grid = (T/16, B), block = 32.
// ---------------------------------------------------------------------------
__global__ __launch_bounds__(32) void attn_kernel(
    const __bf16* __restrict__ Qb, const __bf16* __restrict__ Kb,
    const __bf16* __restrict__ Vt, float* __restrict__ Out) {
  __shared__ __bf16 Plds[16 * 64];   // one wave per block -> private slice

  const int qt = blockIdx.x, b = blockIdx.y;
  const int lane = threadIdx.x & 31, half = lane >> 4, l16 = lane & 15;
  const int t0 = qt * 16;
  const float scale = 0.03125f;      // C^-0.5 = 1024^-0.5 (matches reference)

  // Q tile as 4 A-fragments (16x32 each over the H=128 dim).
  const __bf16* qbase = Qb + ((size_t)b * TT + t0 + l16) * HH + 16 * half;
  v16bf Aq[4];
#pragma unroll
  for (int c = 0; c < 4; ++c) {
    v8bf lo = *(const v8bf*)(qbase + 32 * c);
    v8bf hi = *(const v8bf*)(qbase + 32 * c + 8);
#pragma unroll
    for (int i = 0; i < 8; ++i) { Aq[c][i] = lo[i]; Aq[c][i + 8] = hi[i]; }
  }

  float mrow[8], lrow[8];
  v8f O[8];
#pragma unroll
  for (int r = 0; r < 8; ++r) {
    mrow[r] = -INFINITY;
    lrow[r] = 0.f;
#pragma unroll
    for (int c = 0; c < 8; ++c) O[c][r] = 0.f;
  }

  const __bf16* Kbase = Kb + (size_t)b * TT * HH;
  const __bf16* Vbase = Vt + (size_t)b * HH * TT;

  for (int j = 0; j < t0 + 16; j += 64) {
    // Number of active (not fully causal-masked) 16-key subtiles: 1..4.
    int na = (t0 + 16 - j + 15) >> 4;
    if (na > 4) na = 4;

    v8f S[4];
#pragma unroll
    for (int s = 0; s < 4; ++s)
#pragma unroll
      for (int r = 0; r < 8; ++r) S[s][r] = 0.f;

    // S = Q K^T with K fragments double-buffered across subtiles.
    const __bf16* kp = Kbase + (size_t)(j + l16) * HH + 16 * half;
    v16bf Bk0[4], Bk1[4];
    load_k4(Bk0, kp, 0);
    if (na > 1) load_k4(Bk1, kp, 1);
    S[0] = wmma4(S[0], Aq, Bk0);
    if (na > 1) {
      if (na > 2) load_k4(Bk0, kp, 2);
      S[1] = wmma4(S[1], Aq, Bk1);
    }
    if (na > 2) {
      if (na > 3) load_k4(Bk1, kp, 3);
      S[2] = wmma4(S[2], Aq, Bk0);
    }
    if (na > 3) S[3] = wmma4(S[3], Aq, Bk1);

    // Start the g=0 V-fragment loads now: independent of softmax, their
    // latency hides under the exp/shfl VALU work below.
    const __bf16* vb = Vbase + (size_t)l16 * TT + j + 16 * half;
    v16bf Bv[8];
    load_v8(Bv, vb);

    if (j + 64 < t0 + 16) {
      __builtin_prefetch((const void*)(kp + 64 * HH), 0, 1);
      __builtin_prefetch((const void*)(vb + 64), 0, 1);
    }

    // Online softmax. Row m = r + 8*half lives in VGPR r across a 16-lane
    // half; shfl_xor masks 1..8 stay within the half on wave32.
#pragma unroll
    for (int r = 0; r < 8; ++r) {
      const int m = r + 8 * half;
      const int qrow = t0 + m;
      float sv0 = (j + l16 <= qrow) ? S[0][r] * scale : -INFINITY;
      float sv1 = (na > 1 && j + 16 + l16 <= qrow) ? S[1][r] * scale : -INFINITY;
      float sv2 = (na > 2 && j + 32 + l16 <= qrow) ? S[2][r] * scale : -INFINITY;
      float sv3 = (na > 3 && j + 48 + l16 <= qrow) ? S[3][r] * scale : -INFINITY;
      float t = fmaxf(fmaxf(sv0, sv1), fmaxf(sv2, sv3));
#pragma unroll
      for (int d = 1; d < 16; d <<= 1) t = fmaxf(t, __shfl_xor(t, d, 32));
      const float mnew = fmaxf(mrow[r], t);
      const float corr = __expf(mrow[r] - mnew);
      const float p0 = __expf(sv0 - mnew);   // masked lanes -> exp(-inf) = 0
      const float p1 = __expf(sv1 - mnew);
      const float p2 = __expf(sv2 - mnew);
      const float p3 = __expf(sv3 - mnew);
      float rs = (p0 + p1) + (p2 + p3);
#pragma unroll
      for (int d = 1; d < 16; d <<= 1) rs += __shfl_xor(rs, d, 32);
      lrow[r] = lrow[r] * corr + rs;
      mrow[r] = mnew;
#pragma unroll
      for (int c = 0; c < 8; ++c) O[c][r] *= corr;
      // Stash P row-major in LDS for the C->A fragment transpose.
      Plds[m * 64 + l16]      = (__bf16)p0;
      Plds[m * 64 + 16 + l16] = (__bf16)p1;
      Plds[m * 64 + 32 + l16] = (__bf16)p2;
      Plds[m * 64 + 48 + l16] = (__bf16)p3;
    }

    asm volatile("s_wait_dscnt 0x0" ::: "memory");

    // Re-read P as two A fragments (16x32 each): lane l16 = row m, two 8-elem
    // contiguous K runs at 32g + 16*half and 32g + 16*half + 8.
    v16bf Ap[2];
#pragma unroll
    for (int g = 0; g < 2; ++g) {
      v8bf plo = *(const v8bf*)(&Plds[l16 * 64 + 32 * g + 16 * half]);
      v8bf phi = *(const v8bf*)(&Plds[l16 * 64 + 32 * g + 16 * half + 8]);
#pragma unroll
      for (int i = 0; i < 8; ++i) { Ap[g][i] = plo[i]; Ap[g][i + 8] = phi[i]; }
    }

    // O += P @ V. g=0 fragments are already resident; g=1 (keys j+32..j+63)
    // only when that half of P is nonzero.
    wmma8(O, Ap[0], Bv);
    if (na > 2) {
      load_v8(Bv, vb + 32);
      wmma8(O, Ap[1], Bv);
    }
  }

  // Normalize and store fp32 output [B,T,H].
  float* orow = Out + ((size_t)b * TT + t0) * HH;
#pragma unroll
  for (int r = 0; r < 8; ++r) {
    const float inv = 1.f / lrow[r];
#pragma unroll
    for (int c = 0; c < 8; ++c)
      orow[(size_t)(r + 8 * half) * HH + 16 * c + l16] = O[c][r] * inv;
  }
}

// ---------------------------------------------------------------------------
extern "C" void kernel_launch(void* const* d_in, const int* in_sizes, int n_in,
                              void* d_out, int out_size, void* d_ws, size_t ws_size,
                              hipStream_t stream) {
  (void)in_sizes; (void)n_in; (void)out_size; (void)ws_size;
  const float* X  = (const float*)d_in[0];
  const float* Wk = (const float*)d_in[1];
  const float* Wq = (const float*)d_in[2];
  const float* Wv = (const float*)d_in[3];
  float* Out = (float*)d_out;

  // Workspace layout (bytes):
  //   Wtq 0..256K, Wtk 256K..512K, Wtv 512K..768K   (128x1024 bf16 each)
  //   Qb  768K..+4M, Kb +4M..+8M                    (16384x128 bf16 each)
  //   Vt  +8M..+12M                                 (8x128x2048 bf16)
  char* ws = (char*)d_ws;
  __bf16* Wtq = (__bf16*)(ws);
  __bf16* Wtk = (__bf16*)(ws + 262144);
  __bf16* Wtv = (__bf16*)(ws + 524288);
  __bf16* Qb  = (__bf16*)(ws + 786432);
  __bf16* Kb  = (__bf16*)(ws + 786432 + 4194304);
  __bf16* Vt  = (__bf16*)(ws + 786432 + 2 * 4194304);

  wtrans_kernel<<<512, 256, 0, stream>>>(Wq, Wtq);
  wtrans_kernel<<<512, 256, 0, stream>>>(Wk, Wtk);
  wtrans_kernel<<<512, 256, 0, stream>>>(Wv, Wtv);

  proj_kernel<<<dim3(BB * TT / 16, 3), 32, 0, stream>>>(
      X, Wtq, Wtk, Wtv, Qb, Kb, Vt);

  attn_kernel<<<dim3(TT / 16, BB), 32, 0, stream>>>(Qb, Kb, Vt, Out);
}
